// ScaledDotProductAttention_2319282339925
// MI455X (gfx1250) — matrix operand order repaired and checked by reading, not compiled, not run on hardware
//
#include <hip/hip_runtime.h>
#include <hip/hip_bf16.h>
#include <math.h>

// Scaled dot-product attention, causal, B=2 H=16 S=2048 D=64, fp32 in/out.
// Outputs: context [B,H,S,D] then attn [B,H,S,S], concatenated in d_out.
//
// Fused one-pass kernel: HBM-bound on the 537MB attn write (~23us @ 23.3TB/s).
// K/V (32MiB total) are kept L2-resident across q-tile re-reads; the attn
// stream-out uses non-temporal stores so it does not evict them.

typedef __attribute__((ext_vector_type(16))) _Float16 v16h;
typedef __attribute__((ext_vector_type(8)))  float    v8f;
typedef __attribute__((ext_vector_type(4)))  float    v4f;   // native vec for NT stores

#define AB 2
#define AH 16
#define AS 2048
#define AD 64
#define QT 16          // query rows per workgroup
#define NTHREADS 256   // 8 waves (wave32)

// dynamic LDS layout (bytes)
#define OFF_SCORES 0                        // 16*2048 f32  = 131072
#define OFF_Q      131072                   // 16*64 f16    = 2048
#define OFF_K      133120                   // 8 * 16*64 f16 = 16384
#define OFF_V      149504                   // 8 * 64*32 f16 = 32768
#define OFF_CTX    182272                   // 16*64 f32    = 4096
#define OFF_RED    186368                   // 256 f32      = 1024
#define OFF_RMAX   187392                   // 16 f32
#define OFF_RSUM   187456                   // 16 f32
#define SMEM_BYTES 187520

__global__ __launch_bounds__(NTHREADS) void
sdpa_causal_fused_kernel(const float* __restrict__ q,
                         const float* __restrict__ k,
                         const float* __restrict__ v,
                         float* __restrict__ ctx_out,
                         float* __restrict__ attn_out)
{
    extern __shared__ char smem[];
    float*    sScores = (float*)(smem + OFF_SCORES);     // [16][2048]
    _Float16* sQ      = (_Float16*)(smem + OFF_Q);       // [16][64]
    _Float16* sK      = (_Float16*)(smem + OFF_K);       // per-wave [16][64]
    _Float16* sV      = (_Float16*)(smem + OFF_V);       // per-wave [64][32] (transposed)
    float*    sCtx    = (float*)(smem + OFF_CTX);        // [16][64]
    float*    sRed    = (float*)(smem + OFF_RED);        // [256]
    float*    sRowMax = (float*)(smem + OFF_RMAX);       // [16]
    float*    sRowSum = (float*)(smem + OFF_RSUM);       // [16]

    const int bh   = blockIdx.y;               // 0..31
    const int q0   = blockIdx.x * QT;          // query tile start
    const int tid  = threadIdx.x;
    const int lane = tid & 31;
    const int wave = tid >> 5;
    const int half = lane >> 4;                // lane group (0: lanes 0-15, 1: 16-31)
    const int l16  = lane & 15;

    const float* Qg = q + ((size_t)bh * AS + q0) * AD;
    const float* Kg = k + (size_t)bh * AS * AD;
    const float* Vg = v + (size_t)bh * AS * AD;

    const float NEG = -3.0e38f;

    // ---------------- Phase 0: init LDS ----------------
    {
        float4 n4; n4.x = NEG; n4.y = NEG; n4.z = NEG; n4.w = NEG;
        float4* s4 = (float4*)sScores;
        #pragma unroll 4
        for (int i = tid; i < QT * AS / 4; i += NTHREADS) s4[i] = n4;
        for (int i = tid; i < QT * AD; i += NTHREADS) sCtx[i] = 0.0f;
        // Q tile f32 -> f16 (16*64 = 256 float4, one per thread)
        float4 f = ((const float4*)Qg)[tid];
        int b4 = tid * 4;
        sQ[b4 + 0] = (_Float16)f.x;
        sQ[b4 + 1] = (_Float16)f.y;
        sQ[b4 + 2] = (_Float16)f.z;
        sQ[b4 + 3] = (_Float16)f.w;
    }
    __syncthreads();

    // ---------------- Phase 1: scores = scale * Q K^T, causal masked --------
    // A fragment (Q rows), built once per wave. 16-bit A 16x32 layout:
    // lane holds row m=l16; VGPR v (pair): v<4 -> K = half*8 + 2v,
    // v>=4 -> K = 16 + half*8 + 2(v-4). Two d-slices of 32 cover D=64.
    v16h aQ[2];
    #pragma unroll
    for (int dsl = 0; dsl < 2; ++dsl) {
        #pragma unroll
        for (int vv = 0; vv < 8; ++vv) {
            int kk = (vv < 4) ? (half * 8 + 2 * vv) : (16 + half * 8 + 2 * (vv - 4));
            int d  = dsl * 32 + kk;
            aQ[dsl][2 * vv]     = sQ[l16 * AD + d];
            aQ[dsl][2 * vv + 1] = sQ[l16 * AD + d + 1];
        }
    }

    const int qEnd   = q0 + QT - 1;
    const int numKT  = (qEnd >> 4) + 1;          // causal: k-tiles with any valid col
    _Float16* myK = sK + wave * (16 * AD);

    for (int kt = wave; kt < numKT; kt += 8) {
        // prefetch next K tile owned by this wave (gfx1250 global_prefetch_b8)
        if (kt + 8 < numKT)
            __builtin_prefetch(Kg + (size_t)(kt + 8) * 16 * AD, 0, 0);

        // stage K tile [16 keys][64 d] as f16 (coalesced float4 reads)
        const float4* Ksrc = (const float4*)(Kg + (size_t)kt * 16 * AD);
        #pragma unroll
        for (int i = lane; i < 256; i += 32) {
            float4 f = Ksrc[i];
            int b4 = i * 4;
            myK[b4 + 0] = (_Float16)f.x;
            myK[b4 + 1] = (_Float16)f.y;
            myK[b4 + 2] = (_Float16)f.z;
            myK[b4 + 3] = (_Float16)f.w;
        }
        // B fragment: 16-bit B 32x16 layout: lane holds col n=l16,
        // VGPR v holds K rows (v<4: half*8 + 2v, v>=4: 16 + half*8 + 2(v-4); +1). B[d][n] = K[n][d] -> pair
        // contiguous in d inside myK.
        v8f c = {};
        #pragma unroll
        for (int dsl = 0; dsl < 2; ++dsl) {
            v16h bK;
            #pragma unroll
            for (int vv = 0; vv < 8; ++vv) {
                int d = dsl * 32 + ((vv < 4) ? (half * 8 + 2 * vv) : (16 + half * 8 + 2 * (vv - 4)));
                bK[2 * vv]     = myK[l16 * AD + d];
                bK[2 * vv + 1] = myK[l16 * AD + d + 1];
            }
            c = __builtin_amdgcn_wmma_f32_16x16x32_f16(
                    false, aQ[dsl], false, bK, (short)0, c, false, false);
        }
        // scale + causal mask + scatter into LDS score buffer
        const int gk = kt * 16 + l16;
        #pragma unroll
        for (int r = 0; r < 8; ++r) {
            int m = r + half * 8;
            float val = c[r] * 0.125f;             // 1/sqrt(64)
            if (gk > q0 + m) val = NEG;
            sScores[m * AS + kt * 16 + l16] = val;
        }
    }
    __syncthreads();

    // ---------------- Phase 2: row softmax reductions ----------------
    {
        const int row = tid >> 4;       // 0..15
        const int sub = tid & 15;
        const float* rowp = sScores + row * AS;
        float mx = NEG;
        for (int j = sub * 128; j < sub * 128 + 128; ++j) mx = fmaxf(mx, rowp[j]);
        sRed[tid] = mx;
        __syncthreads();
        if (sub == 0) {
            float m2 = sRed[tid];
            #pragma unroll
            for (int t = 1; t < 16; ++t) m2 = fmaxf(m2, sRed[tid + t]);
            sRowMax[row] = m2;
        }
        __syncthreads();
        const float rm = sRowMax[row];
        float sum = 0.0f;
        for (int j = sub * 128; j < sub * 128 + 128; ++j) sum += __expf(rowp[j] - rm);
        sRed[tid] = sum;
        __syncthreads();
        if (sub == 0) {
            float t2 = sRed[tid];
            #pragma unroll
            for (int t = 1; t < 16; ++t) t2 += sRed[tid + t];
            sRowSum[row] = 1.0f / t2;
        }
        __syncthreads();
    }

    // ---------------- Phase 3: normalize in LDS + stream attn to HBM --------
    // Non-temporal stores: the 537MB attn stream is ~8x L2 capacity; NT (rinse)
    // keeps the K/V working set resident in L2 for the other workgroups.
    {
        float* attnBase = attn_out + ((size_t)bh * AS + q0) * AS;  // 16 contiguous rows
        v4f* s4 = (v4f*)sScores;
        v4f* a4 = (v4f*)attnBase;
        #pragma unroll 4
        for (int i = tid; i < QT * AS / 4; i += NTHREADS) {
            int r = i >> 9;                      // (i*4)/2048
            float rm = sRowMax[r], is = sRowSum[r];
            v4f sv = s4[i];
            v4f p;
            p.x = __expf(sv.x - rm) * is;
            p.y = __expf(sv.y - rm) * is;
            p.z = __expf(sv.z - rm) * is;
            p.w = __expf(sv.w - rm) * is;
            s4[i] = p;                               // keep probs for P@V
            __builtin_nontemporal_store(p, &a4[i]);  // b128 store, TH=NT
        }
    }
    __syncthreads();

    // ---------------- Phase 4: context = P @ V ----------------
    {
        const int numCh = (qEnd >> 5) + 1;       // 32-key chunks with any mass
        _Float16* myV = sV + wave * (AD * 32);   // transposed [d][k]
        v8f acc[4] = { {}, {}, {}, {} };

        for (int c = wave; c < numCh; c += 8) {
            const int k0 = c * 32;
            if (c + 8 < numCh)
                __builtin_prefetch(Vg + (size_t)(k0 + 256) * AD, 0, 0);
            // stage V chunk [32 keys][64 d] -> LDS transposed f16 [d][k]
            const float4* Vsrc = (const float4*)(Vg + (size_t)k0 * AD);
            #pragma unroll
            for (int i = lane; i < 512; i += 32) {
                float4 f = Vsrc[i];
                int kk = i >> 4;                 // key 0..31
                int d  = (i & 15) * 4;
                myV[(d + 0) * 32 + kk] = (_Float16)f.x;
                myV[(d + 1) * 32 + kk] = (_Float16)f.y;
                myV[(d + 2) * 32 + kk] = (_Float16)f.z;
                myV[(d + 3) * 32 + kk] = (_Float16)f.w;
            }
            // A = P tile 16x32 (probs f32 -> f16)
            v16h aP;
            #pragma unroll
            for (int vv = 0; vv < 8; ++vv) {
                int kk = (vv < 4) ? (half * 8 + 2 * vv) : (16 + half * 8 + 2 * (vv - 4));
                aP[2 * vv]     = (_Float16)sScores[l16 * AS + k0 + kk];
                aP[2 * vv + 1] = (_Float16)sScores[l16 * AS + k0 + kk + 1];
            }
            // B = V slice 32x16 per 16-wide N-tile of D=64; pair rows (k, k+1)
            // contiguous in transposed myV.
            #pragma unroll
            for (int nt = 0; nt < 4; ++nt) {
                v16h bV;
                #pragma unroll
                for (int vv = 0; vv < 8; ++vv) {
                    int krow = (vv < 4) ? (half * 8 + 2 * vv) : (16 + half * 8 + 2 * (vv - 4));
                    int d    = nt * 16 + l16;
                    bV[2 * vv]     = myV[d * 32 + krow];
                    bV[2 * vv + 1] = myV[d * 32 + krow + 1];
                }
                acc[nt] = __builtin_amdgcn_wmma_f32_16x16x32_f16(
                              false, aP, false, bV, (short)0, acc[nt], false, false);
            }
        }
        // cross-wave reduction via LDS float atomics (ds_add_f32)
        #pragma unroll
        for (int nt = 0; nt < 4; ++nt) {
            #pragma unroll
            for (int r = 0; r < 8; ++r) {
                int m = r + half * 8;
                int d = nt * 16 + l16;
                atomicAdd(&sCtx[m * AD + d], acc[nt][r]);
            }
        }
        __syncthreads();
        // write context tile (16*64 = 256 float4, one per thread); written once,
        // never re-read by the kernel -> non-temporal.
        float* ctxBase = ctx_out + ((size_t)bh * AS + q0) * AD;
        v4f cv = ((v4f*)sCtx)[tid];
        __builtin_nontemporal_store(cv, &((v4f*)ctxBase)[tid]);
    }
}

extern "C" void kernel_launch(void* const* d_in, const int* in_sizes, int n_in,
                              void* d_out, int out_size, void* d_ws, size_t ws_size,
                              hipStream_t stream) {
    (void)in_sizes; (void)n_in; (void)d_ws; (void)ws_size; (void)out_size;
    const float* q = (const float*)d_in[0];
    const float* k = (const float*)d_in[1];
    const float* v = (const float*)d_in[2];
    // d_in[3] is the boolean causal mask; computed analytically in-kernel.

    float* ctx_out  = (float*)d_out;                                  // [B,H,S,D]
    float* attn_out = ctx_out + (size_t)AB * AH * AS * AD;            // [B,H,S,S]

    // Allow >64KB dynamic LDS (CDNA5 WGP supports 320KB per workgroup).
    // Called unconditionally: deterministic, host-side, capture-safe.
    (void)hipFuncSetAttribute((const void*)sdpa_causal_fused_kernel,
                              hipFuncAttributeMaxDynamicSharedMemorySize, SMEM_BYTES);

    dim3 grid(AS / QT, AB * AH);   // (128, 32)
    sdpa_causal_fused_kernel<<<grid, NTHREADS, SMEM_BYTES, stream>>>(
        q, k, v, ctx_out, attn_out);
}
